// MGNN_18717467476640
// MI455X (gfx1250) — compile-verified, matrix operand order checked by reading, not run on hardware
//
#include <hip/hip_runtime.h>

// ---------------------------------------------------------------------------
// MGNN fused forward for MI455X (gfx1250, wave32, WMMA bf16).
// B=2, N=4096, D_IN=256, D_OUT=64, K=3, DEGREE=3, BANDWIDTH=512.
// Pipeline:
//   prep:  x -> bf16 (+ transposed copy), row sq-norms, weights -> bf16
//          transposed
//   hop x2 (fused, flash-style): S = x_i.x_j^T via WMMA, elementwise
//          spline/wavelet kernels in f32, LDS relayout, second WMMA chain
//          h_next = K . h_prev  (both kernel types per pass).  All B-operands
//          are read from K-contiguous (transposed) layouts so every fragment
//          load is a global_load_b128.
//   head:  out = relu( x@Wk0 + h1@Wk1 + h2@Wk2 (per half) + x@W_lin + b_lin )
// ---------------------------------------------------------------------------

#define BN 2
#define NN 4096
#define DD 256
#define DO 64
#define LDSP 136   // padded row stride (128 + 8 bf16) to avoid LDS bank conflicts

typedef __attribute__((ext_vector_type(16))) __bf16 v16bf;
typedef __attribute__((ext_vector_type(8)))  float  v8f;

union Frag {
  v16bf v;
  unsigned int u[8];
};

__device__ __forceinline__ v8f zero8() {
  v8f z;
  #pragma unroll
  for (int i = 0; i < 8; ++i) z[i] = 0.f;
  return z;
}

// K-offset of VGPR pair v for a 16-bit 16xK fragment (ISA 7.12.2), half = lane>=16.
__device__ __forceinline__ int koff(int v, int h) {
  return (v < 4) ? (2 * v + 8 * h) : (16 + 2 * (v - 4) + 8 * h);
}

// Fragment whose consecutive K elements are contiguous in memory.
// rowptr points at element (row, kbase); 4-byte (in fact 16-byte) aligned.
__device__ __forceinline__ v16bf load_frag_rowmajor(const __bf16* rowptr, int h) {
  Frag f;
  #pragma unroll
  for (int v = 0; v < 8; ++v)
    f.u[v] = *reinterpret_cast<const unsigned int*>(rowptr + koff(v, h));
  return f.v;
}

__device__ __forceinline__ v8f wmma_bf16(v16bf a, v16bf b, v8f c) {
  return __builtin_amdgcn_wmma_f32_16x16x32_bf16(false, a, false, b,
                                                 (short)0, c, false, false);
}

// Cubic Cox-de Boor B-spline N_{0,3}(t) on uniform knots [0,.25,.5,.75,1],
// matching the reference's half-open indicator intervals exactly.
__device__ __forceinline__ float bspline3(float t) {
  float b0 = (t >= 0.00f && t < 0.25f) ? 1.f : 0.f;
  float b1 = (t >= 0.25f && t < 0.50f) ? 1.f : 0.f;
  float b2 = (t >= 0.50f && t < 0.75f) ? 1.f : 0.f;
  float b3 = (t >= 0.75f && t < 1.00f) ? 1.f : 0.f;
  float c0 = (t - 0.00f) * 4.f * b0 + (0.50f - t) * 4.f * b1;
  float c1 = (t - 0.25f) * 4.f * b1 + (0.75f - t) * 4.f * b2;
  float c2 = (t - 0.50f) * 4.f * b2 + (1.00f - t) * 4.f * b3;
  float d0 = (t - 0.00f) * 2.f * c0 + (0.75f - t) * 2.f * c1;
  float d1 = (t - 0.25f) * 2.f * c1 + (1.00f - t) * 2.f * c2;
  return (t - 0.00f) * (1.f / 0.75f) * d0 + (1.00f - t) * (1.f / 0.75f) * d1;
}

// ------------------------------ prep kernels -------------------------------

__global__ void convert_f32_bf16(const float* __restrict__ s,
                                 __bf16* __restrict__ d, int n) {
  int i = blockIdx.x * blockDim.x + threadIdx.x;
  if (i < n) d[i] = (__bf16)s[i];
}

// s: [R][C] f32  ->  d: [C][R] bf16   (small weight matrices)
__global__ void convert_transpose_f32_bf16(const float* __restrict__ s,
                                           __bf16* __restrict__ d,
                                           int R, int C) {
  int i = blockIdx.x * blockDim.x + threadIdx.x;  // output-linear [c][r]
  if (i >= R * C) return;
  int c = i / R, r = i % R;
  d[i] = (__bf16)s[(size_t)r * C + c];
}

__global__ void rowsq_kernel(const float* __restrict__ x,
                             float* __restrict__ sq, int rows) {
  int wid  = (blockIdx.x * blockDim.x + threadIdx.x) >> 5;
  int lane = threadIdx.x & 31;
  if (wid >= rows) return;
  const float* r = x + (size_t)wid * DD;
  float s = 0.f;
  #pragma unroll
  for (int k = 0; k < DD / 32; ++k) {
    float v = r[lane + 32 * k];
    s += v * v;
  }
  #pragma unroll
  for (int off = 16; off; off >>= 1) s += __shfl_xor(s, off, 32);
  if (lane == 0) sq[wid] = s;
}

// LDS-tiled bf16 transpose: src [B][NN][DD] -> dst [B][DD][NN]
__global__ __launch_bounds__(256) void transpose_bf16(
    const __bf16* __restrict__ src, __bf16* __restrict__ dst) {
  __shared__ __bf16 tile[32][33];
  const int tilesPerBatch = (NN / 32) * (DD / 32);
  int blk = blockIdx.x;
  int b   = blk / tilesPerBatch;
  int r   = blk % tilesPerBatch;
  int n0  = (r / (DD / 32)) * 32;
  int d0  = (r % (DD / 32)) * 32;
  int tx  = threadIdx.x & 31;
  int ty  = threadIdx.x >> 5;  // 0..7
  #pragma unroll
  for (int p = 0; p < 4; ++p)
    tile[ty + 8 * p][tx] = src[((size_t)b * NN + n0 + ty + 8 * p) * DD + d0 + tx];
  __syncthreads();
  #pragma unroll
  for (int p = 0; p < 4; ++p)
    dst[((size_t)b * DD + d0 + ty + 8 * p) * NN + n0 + tx] = tile[tx][ty + 8 * p];
}

// ------------------------------- hop kernel --------------------------------
// One workgroup (4 waves) owns a 16-row tile of one batch and sweeps all j.
// Phase 1: wave w builds S for j-subchunk [j0+32w, j0+32w+32) via bf16 WMMA,
//          turns it into bspline/wavelet kernel tiles, writes bf16 to LDS.
// Phase 2: every wave consumes the shared 16x128 LDS kernel tiles as A-frags
//          and accumulates its 64-column slice of H_b and H_w, reading the
//          propagated features from the transposed [B][DD][NN] layout so
//          B-fragments are K-contiguous (global_load_b128).

__global__ __launch_bounds__(128) void hop_kernel(
    const __bf16* __restrict__ xb,    // [B][NN][DD] (Gram operands)
    const float* __restrict__ sq,
    const __bf16* __restrict__ srcBT, // [B][DD][NN] transposed
    const __bf16* __restrict__ srcWT, // [B][DD][NN] transposed
    __bf16* __restrict__ dstB,        // [B][NN][DD]
    __bf16* __restrict__ dstW) {
  __shared__ __bf16 bkLds[16 * LDSP];
  __shared__ __bf16 wkLds[16 * LDSP];

  const int lane = threadIdx.x & 31;
  const int w    = threadIdx.x >> 5;
  const int hh   = lane >> 4;
  const int ln   = lane & 15;
  const int blk  = blockIdx.x;
  const int b    = blk >> 8;           // NN/16 == 256 row tiles per batch
  const int i0   = (blk & 255) << 4;

  const __bf16* xbat = xb + (size_t)b * NN * DD;
  const float*  sqb  = sq + (size_t)b * NN;

  float si[8];
  #pragma unroll
  for (int r = 0; r < 8; ++r) si[r] = sqb[i0 + r + 8 * hh];

  v8f Cb[4], Cw[4];
  #pragma unroll
  for (int t = 0; t < 4; ++t) { Cb[t] = zero8(); Cw[t] = zero8(); }

  for (int j0 = 0; j0 < NN; j0 += 128) {
    // ---- phase 1: Gram sub-tiles for this wave ----
    const int jb = j0 + w * 32;
    v8f S0 = zero8(), S1 = zero8();
    #pragma unroll
    for (int d0 = 0; d0 < DD; d0 += 32) {
      v16bf af  = load_frag_rowmajor(xbat + (size_t)(i0 + ln) * DD + d0, hh);
      v16bf bf0 = load_frag_rowmajor(xbat + (size_t)(jb + ln) * DD + d0, hh);
      v16bf bf1 = load_frag_rowmajor(xbat + (size_t)(jb + 16 + ln) * DD + d0, hh);
      S0 = wmma_bf16(af, bf0, S0);
      S1 = wmma_bf16(af, bf1, S1);
    }

    // ---- elementwise: adjacency -> spline / wavelet kernels, to LDS ----
    #pragma unroll
    for (int tt = 0; tt < 2; ++tt) {
      v8f S = tt ? S1 : S0;
      int jg   = jb + tt * 16 + ln;
      float sj = sqb[jg];
      #pragma unroll
      for (int r = 0; r < 8; ++r) {
        float d2  = si[r] + sj - 2.f * S[r];
        d2        = fmaxf(d2, 0.f);
        float adj = __expf(-d2 * (1.f / 512.f));
        float bk  = bspline3(adj);
        float wk  = bspline3(2.f * adj) - bspline3(2.f * adj - 1.f);
        int m  = r + 8 * hh;
        int kk = w * 32 + tt * 16 + ln;
        bkLds[m * LDSP + kk] = (__bf16)bk;
        wkLds[m * LDSP + kk] = (__bf16)wk;
      }
    }
    __syncthreads();

    // ---- phase 2: C += K_tile(16x128) @ src(128 x my 64 cols) ----
    #pragma unroll
    for (int kc = 0; kc < 4; ++kc) {
      v16bf aB = load_frag_rowmajor(&bkLds[ln * LDSP + kc * 32], hh);
      v16bf aW = load_frag_rowmajor(&wkLds[ln * LDSP + kc * 32], hh);
      #pragma unroll
      for (int ct = 0; ct < 4; ++ct) {
        // B element (k, n) = srcT[b][w*64 + ct*16 + n][j0 + kc*32 + k]
        const size_t rowbase =
            ((size_t)b * DD + w * 64 + ct * 16 + ln) * NN + j0 + kc * 32;
        v16bf bfB = load_frag_rowmajor(srcBT + rowbase, hh);
        v16bf bfW = load_frag_rowmajor(srcWT + rowbase, hh);
        Cb[ct] = wmma_bf16(aB, bfB, Cb[ct]);
        Cw[ct] = wmma_bf16(aW, bfW, Cw[ct]);
      }
    }
    __syncthreads();
  }

  #pragma unroll
  for (int ct = 0; ct < 4; ++ct) {
    #pragma unroll
    for (int r = 0; r < 8; ++r) {
      int m = r + 8 * hh;
      size_t o = ((size_t)b * NN + i0 + m) * DD + w * 64 + ct * 16 + ln;
      dstB[o] = (__bf16)Cb[ct][r];
      dstW[o] = (__bf16)Cw[ct][r];
    }
  }
}

// ------------------------------- head kernel -------------------------------
// One wave per (batch, 16-row tile, half): 16x64 output block.
//   half 0 (cols 0..63)  : x@Wk0 + h1b@Wk1 + h2b@Wk2 + x@Wlin[:, 0:64]
//   half 1 (cols 64..127): x@Wk0 + h1w@Wk1 + h2w@Wk2 + x@Wlin[:, 64:128]
// Weights are pre-transposed so B-fragments are K-contiguous.

__global__ __launch_bounds__(256) void head_kernel(
    const __bf16* __restrict__ xb,
    const __bf16* __restrict__ h1b, const __bf16* __restrict__ h1w,
    const __bf16* __restrict__ h2b, const __bf16* __restrict__ h2w,
    const __bf16* __restrict__ wkT,   // [3][64][256]  (o-major)
    const __bf16* __restrict__ wlT,   // [128][256]    (o-major)
    const float* __restrict__ b_lin,  // [128]
    float* __restrict__ out) {        // [B][N][128]
  int unit = blockIdx.x * 8 + (threadIdx.x >> 5);  // 0..1023
  int lane = threadIdx.x & 31;
  int hh = lane >> 4, ln = lane & 15;
  int t  = unit & 1;
  int rt = (unit >> 1) & 255;
  int b  = unit >> 9;
  int i0 = rt * 16;

  const __bf16* h1 = t ? h1w : h1b;
  const __bf16* h2 = t ? h2w : h2b;

  v8f C[4];
  #pragma unroll
  for (int ct = 0; ct < 4; ++ct) C[ct] = zero8();

  const size_t rowoff = ((size_t)b * NN + i0 + ln) * DD;
  #pragma unroll
  for (int d0 = 0; d0 < DD; d0 += 32) {
    v16bf ax = load_frag_rowmajor(xb + rowoff + d0, hh);
    v16bf a1 = load_frag_rowmajor(h1 + rowoff + d0, hh);
    v16bf a2 = load_frag_rowmajor(h2 + rowoff + d0, hh);
    #pragma unroll
    for (int ct = 0; ct < 4; ++ct) {
      int oc = ct * 16;
      // B element (k=d, n=o) = W[d0+k][oc+n] = WT[oc+n][d0+k]
      v16bf w0 = load_frag_rowmajor(wkT + (size_t)(0 * DO + oc + ln) * DD + d0, hh);
      v16bf w1 = load_frag_rowmajor(wkT + (size_t)(1 * DO + oc + ln) * DD + d0, hh);
      v16bf w2 = load_frag_rowmajor(wkT + (size_t)(2 * DO + oc + ln) * DD + d0, hh);
      v16bf wl = load_frag_rowmajor(wlT + (size_t)(t * 64 + oc + ln) * DD + d0, hh);
      C[ct] = wmma_bf16(ax, w0, C[ct]);
      C[ct] = wmma_bf16(a1, w1, C[ct]);
      C[ct] = wmma_bf16(a2, w2, C[ct]);
      C[ct] = wmma_bf16(ax, wl, C[ct]);
    }
  }

  #pragma unroll
  for (int ct = 0; ct < 4; ++ct) {
    int col = t * 64 + ct * 16 + ln;
    float bias = b_lin[col];
    #pragma unroll
    for (int r = 0; r < 8; ++r) {
      int m = r + 8 * hh;
      float v = C[ct][r] + bias;
      out[((size_t)b * NN + i0 + m) * 128 + col] = fmaxf(v, 0.f);
    }
  }
}

// ------------------------------ host launcher ------------------------------

extern "C" void kernel_launch(void* const* d_in, const int* in_sizes, int n_in,
                              void* d_out, int out_size, void* d_ws, size_t ws_size,
                              hipStream_t stream) {
  const float* x    = (const float*)d_in[0];  // [2,4096,256]
  const float* Wk   = (const float*)d_in[1];  // [3,256,64]
  const float* Wlin = (const float*)d_in[2];  // [256,128]
  const float* blin = (const float*)d_in[3];  // [128]
  float* out = (float*)d_out;

  char* ws = (char*)d_ws;
  size_t off = 0;
  auto alloc = [&](size_t bytes) -> void* {
    void* p = ws + off;
    off += (bytes + 255) & ~(size_t)255;
    return p;
  };
  const size_t hbytes = (size_t)BN * NN * DD * sizeof(__bf16);  // 4 MiB each
  __bf16* xb   = (__bf16*)alloc(hbytes);
  __bf16* xbT  = (__bf16*)alloc(hbytes);
  __bf16* h1b  = (__bf16*)alloc(hbytes);
  __bf16* h1w  = (__bf16*)alloc(hbytes);
  __bf16* h1bT = (__bf16*)alloc(hbytes);
  __bf16* h1wT = (__bf16*)alloc(hbytes);
  __bf16* h2b  = (__bf16*)alloc(hbytes);
  __bf16* h2w  = (__bf16*)alloc(hbytes);
  float*  sq   = (float*)alloc((size_t)BN * NN * sizeof(float));
  __bf16* wkT  = (__bf16*)alloc((size_t)3 * DD * DO * sizeof(__bf16));
  __bf16* wlT  = (__bf16*)alloc((size_t)DD * 128 * sizeof(__bf16));
  (void)ws_size; (void)in_sizes; (void)n_in; (void)out_size;
  // total workspace ~34 MB

  const int nx = BN * NN * DD;
  const int tilesT = BN * (NN / 32) * (DD / 32);  // 2048 tiles
  convert_f32_bf16<<<(nx + 255) / 256, 256, 0, stream>>>(x, xb, nx);
  // Wk: three [256][64] -> [64][256] transposed bf16
  for (int kk = 0; kk < 3; ++kk)
    convert_transpose_f32_bf16<<<(DD * DO + 255) / 256, 256, 0, stream>>>(
        Wk + (size_t)kk * DD * DO, wkT + (size_t)kk * DO * DD, DD, DO);
  convert_transpose_f32_bf16<<<(DD * 128 + 255) / 256, 256, 0, stream>>>(
      Wlin, wlT, DD, 128);
  rowsq_kernel<<<(BN * NN * 32 + 255) / 256, 256, 0, stream>>>(x, sq, BN * NN);
  transpose_bf16<<<tilesT, 256, 0, stream>>>(xb, xbT);

  // hop 1: h1 = K . x for both kernel types (src = x, transposed layout)
  hop_kernel<<<BN * (NN / 16), 128, 0, stream>>>(xb, sq, xbT, xbT, h1b, h1w);
  transpose_bf16<<<tilesT, 256, 0, stream>>>(h1b, h1bT);
  transpose_bf16<<<tilesT, 256, 0, stream>>>(h1w, h1wT);
  // hop 2: h2 = K . h1
  hop_kernel<<<BN * (NN / 16), 128, 0, stream>>>(xb, sq, h1bT, h1wT, h2b, h2w);

  // head: projections + residual linear + bias + relu
  head_kernel<<<(BN * (NN / 16) * 2) / 8, 256, 0, stream>>>(
      xb, h1b, h1w, h2b, h2w, wkT, wlT, blin, out);
}
